// Model_76673756168624
// MI455X (gfx1250) — compile-verified, matrix operand order checked by reading, not run on hardware
//
#include <hip/hip_runtime.h>

#define NEG_SLOPE 0.2f
#define N_SRC 100000
#define N_DST 50000
#define N_EDGES 3200000
#define NH 4

typedef __attribute__((ext_vector_type(16))) _Float16 v16h;
typedef __attribute__((ext_vector_type(8)))  _Float16 v8h;
typedef __attribute__((ext_vector_type(8)))  float    v8f;

__device__ __forceinline__ float lrelu(float x){ return x >= 0.f ? x : NEG_SLOPE * x; }

// Order-preserving f32 <-> u32 mapping so we can use integer atomicMax for the
// segment max. All real encodings are > 0, so a 0-initialized cell acts as -inf.
__device__ __forceinline__ unsigned enc_f32(float x){
    unsigned u = __float_as_uint(x);
    return (u & 0x80000000u) ? ~u : (u | 0x80000000u);
}
__device__ __forceinline__ float dec_f32(unsigned u){
    return __uint_as_float((u & 0x80000000u) ? (u & 0x7fffffffu) : ~u);
}

// Load a 16x32 (MxK) f16 WMMA A-fragment (or, mirrored, a KxN B-fragment where
// each lane owns one N column) from a row-major f32 matrix, converting to f16.
// ISA layout: lane L row (L&15); K = kb..kb+7 in v[0..7], kb+16..kb+23 in
// v[8..15], kb = (L>>4)*8.
__device__ __forceinline__ v16h load_frag_f32(const float* __restrict__ p, int ld,
                                              int row0, int k0, int lane){
    int r  = row0 + (lane & 15);
    int kb = k0 + ((lane >> 4) << 3);
    const float* q = p + (size_t)r * ld + kb;
    v16h v;
#pragma unroll
    for (int i = 0; i < 8; ++i){
        v[i]   = (_Float16)q[i];
        v[8+i] = (_Float16)q[16+i];
    }
    return v;
}

// ---------------------------------------------------------------------------
// Init workspace (g, segsum zeroed; segmax encoded -inf == 0)
// ---------------------------------------------------------------------------
__global__ void gat_init(float* __restrict__ g, float* __restrict__ segsum,
                         unsigned* __restrict__ segmax){
    int i = blockIdx.x * 256 + threadIdx.x;
    if (i < N_DST * 128) g[i] = 0.f;
    if (i < N_DST * NH){ segsum[i] = 0.f; segmax[i] = 0u; }
}

// ---------------------------------------------------------------------------
// fs / fd projection: out[M,512] (f16) = A[M,128] @ W[512,128]^T + bias
// One wave per 16x16 tile, K=128 in 4 WMMA steps.
// ---------------------------------------------------------------------------
__global__ void gat_gemm512(const float* __restrict__ A, const float* __restrict__ W,
                            const float* __restrict__ bias, _Float16* __restrict__ out,
                            int M){
    int lane = threadIdx.x & 31;
    int wave = blockIdx.x * 8 + (threadIdx.x >> 5);
    int row0 = (wave >> 5) << 4;   // 32 column tiles (512/16)
    int col0 = (wave & 31) << 4;
    if (row0 >= M) return;
    v8f c = {};
#pragma unroll
    for (int kk = 0; kk < 4; ++kk){
        v16h a = load_frag_f32(A, 128, row0, kk * 32, lane);
        v16h b = load_frag_f32(W, 128, col0, kk * 32, lane);
        c = __builtin_amdgcn_wmma_f32_16x16x32_f16(false, a, false, b, (short)0, c, false, false);
    }
    int n  = col0 + (lane & 15);
    int mb = row0 + ((lane >> 4) << 3);
    float bv = bias[n];
#pragma unroll
    for (int r = 0; r < 8; ++r)
        out[(size_t)(mb + r) * 512 + n] = (_Float16)(c[r] + bv);
}

// ---------------------------------------------------------------------------
// Per-edge attention logits + segment max. One wave32 per edge. Lane L owns
// the 16 contiguous features [16L, 16L+16) of the flattened [H=4,128] vector,
// so head = L>>3 and each head's 128-feature dot product reduces across one
// 8-lane group (3 shfl_xor rounds, offsets 1/2/4 stay inside the group).
// Per-lane traffic: 2x b128 (fs) + 2x b128 (fd) + 4x b128 (attn).
// ---------------------------------------------------------------------------
__global__ void gat_logits(const _Float16* __restrict__ fs, const _Float16* __restrict__ fd,
                           const int* __restrict__ esrc, const int* __restrict__ edst,
                           const float* __restrict__ attn, float* __restrict__ exlog,
                           unsigned* __restrict__ segmax){
    int lane = threadIdx.x & 31;
    int e    = blockIdx.x * 8 + (threadIdx.x >> 5);
    int s = esrc[e], d = edst[e];
    const v8h* ps = (const v8h*)(fs + (size_t)s * 512);
    const v8h* pd = (const v8h*)(fd + (size_t)d * 512);
    v8h s0 = ps[lane * 2], s1 = ps[lane * 2 + 1];
    v8h d0 = pd[lane * 2], d1 = pd[lane * 2 + 1];
    const float4* pa = (const float4*)attn;   // attn is [4*128] f32
    float part = 0.f;
#pragma unroll
    for (int q = 0; q < 4; ++q){
        float4 av = pa[lane * 4 + q];
        float v0, v1, v2, v3;
        if (q < 2){
            v0 = (float)s0[q*4+0] + (float)d0[q*4+0];
            v1 = (float)s0[q*4+1] + (float)d0[q*4+1];
            v2 = (float)s0[q*4+2] + (float)d0[q*4+2];
            v3 = (float)s0[q*4+3] + (float)d0[q*4+3];
        } else {
            v0 = (float)s1[(q-2)*4+0] + (float)d1[(q-2)*4+0];
            v1 = (float)s1[(q-2)*4+1] + (float)d1[(q-2)*4+1];
            v2 = (float)s1[(q-2)*4+2] + (float)d1[(q-2)*4+2];
            v3 = (float)s1[(q-2)*4+3] + (float)d1[(q-2)*4+3];
        }
        part += lrelu(v0) * av.x + lrelu(v1) * av.y + lrelu(v2) * av.z + lrelu(v3) * av.w;
    }
#pragma unroll
    for (int off = 1; off < 8; off <<= 1)
        part += __shfl_xor(part, off, 32);
    int h = lane >> 3;                 // head owned by this lane's group
    if ((lane & 7) == 0){
        exlog[(size_t)e * NH + h] = part;
        atomicMax(&segmax[d * NH + h], enc_f32(part));
    }
}

// ---------------------------------------------------------------------------
// ex = exp(logit - segmax[dst]) (in place) and segment sum via atomicAdd.
// ---------------------------------------------------------------------------
__global__ void gat_expsum(const int* __restrict__ edst, const unsigned* __restrict__ segmax,
                           float* __restrict__ exlog, float* __restrict__ segsum){
    int i = blockIdx.x * 256 + threadIdx.x;   // over E*NH, exact multiple
    int e = i >> 2, h = i & 3;
    int d = edst[e];
    float x = __expf(exlog[i] - dec_f32(segmax[d * NH + h]));
    exlog[i] = x;
    atomicAdd(&segsum[d * NH + h], x);
}

// ---------------------------------------------------------------------------
// Message: per edge, per head softmax over 128 features of env*rr*a, summed
// over heads, scatter-added into g[dst,128]. One wave32 per edge; b128 gathers.
// ---------------------------------------------------------------------------
__global__ void gat_message(const float* __restrict__ src_feat, const float* __restrict__ rel_emb,
                            const int* __restrict__ esrc, const int* __restrict__ edst,
                            const int* __restrict__ etype, const float* __restrict__ ex,
                            const float* __restrict__ segsum, float* __restrict__ g){
    int lane = threadIdx.x & 31;
    int e    = blockIdx.x * 8 + (threadIdx.x >> 5);
    int s = esrc[e], d = edst[e], t = etype[e];
    float4 env4 = ((const float4*)(src_feat + (size_t)s * 128))[lane];
    float4 rr4  = ((const float4*)(rel_emb  + (size_t)t * 128))[lane];
    float x[4] = { env4.x * rr4.x, env4.y * rr4.y, env4.z * rr4.z, env4.w * rr4.w };
    float acc[4] = {0.f, 0.f, 0.f, 0.f};
#pragma unroll
    for (int h = 0; h < NH; ++h){
        float a = ex[(size_t)e * NH + h] / segsum[d * NH + h];
        float y[4];
        float mx = -3.402823466e38f;
#pragma unroll
        for (int j = 0; j < 4; ++j){ y[j] = x[j] * a; mx = fmaxf(mx, y[j]); }
#pragma unroll
        for (int off = 16; off > 0; off >>= 1)
            mx = fmaxf(mx, __shfl_xor(mx, off, 32));
        float sum = 0.f;
#pragma unroll
        for (int j = 0; j < 4; ++j){ y[j] = __expf(y[j] - mx); sum += y[j]; }
#pragma unroll
        for (int off = 16; off > 0; off >>= 1)
            sum += __shfl_xor(sum, off, 32);
        float inv = 1.f / sum;
#pragma unroll
        for (int j = 0; j < 4; ++j) acc[j] += y[j] * inv;
    }
    int f = lane * 4;
#pragma unroll
    for (int j = 0; j < 4; ++j)
        atomicAdd(&g[(size_t)d * 128 + f + j], acc[j]);
}

// ---------------------------------------------------------------------------
// out = lrelu((dst+g)@w1^T + b1) + lrelu((dst*g)@w4^T + b4)
// One wave per 16x16 output tile; both GEMMs fused (A fragments built on the
// fly from dst_feat and g; independent accumulators interleave the two WMMAs).
// ---------------------------------------------------------------------------
__global__ void gat_final(const float* __restrict__ dstf, const float* __restrict__ g,
                          const float* __restrict__ w1, const float* __restrict__ b1,
                          const float* __restrict__ w4, const float* __restrict__ b4,
                          float* __restrict__ out){
    int lane = threadIdx.x & 31;
    int wave = blockIdx.x * 8 + (threadIdx.x >> 5);
    int row0 = (wave >> 3) << 4;   // 8 column tiles (128/16)
    int col0 = (wave & 7) << 4;
    int r = row0 + (lane & 15);
    v8f c1 = {}, c2 = {};
#pragma unroll
    for (int kk = 0; kk < 4; ++kk){
        int kb = kk * 32 + ((lane >> 4) << 3);
        const float* pd = dstf + (size_t)r * 128 + kb;
        const float* pg = g    + (size_t)r * 128 + kb;
        v16h a1, a2;
#pragma unroll
        for (int i = 0; i < 8; ++i){
            float dv = pd[i], gv = pg[i];
            a1[i] = (_Float16)(dv + gv);
            a2[i] = (_Float16)(dv * gv);
            dv = pd[16 + i]; gv = pg[16 + i];
            a1[8 + i] = (_Float16)(dv + gv);
            a2[8 + i] = (_Float16)(dv * gv);
        }
        v16h bb1 = load_frag_f32(w1, 128, col0, kk * 32, lane);
        v16h bb2 = load_frag_f32(w4, 128, col0, kk * 32, lane);
        c1 = __builtin_amdgcn_wmma_f32_16x16x32_f16(false, a1, false, bb1, (short)0, c1, false, false);
        c2 = __builtin_amdgcn_wmma_f32_16x16x32_f16(false, a2, false, bb2, (short)0, c2, false, false);
    }
    int n  = col0 + (lane & 15);
    int mb = row0 + ((lane >> 4) << 3);
    float bv1 = b1[n], bv4 = b4[n];
#pragma unroll
    for (int q = 0; q < 8; ++q)
        out[(size_t)(mb + q) * 128 + n] = lrelu(c1[q] + bv1) + lrelu(c2[q] + bv4);
}

// ---------------------------------------------------------------------------
extern "C" void kernel_launch(void* const* d_in, const int* in_sizes, int n_in,
                              void* d_out, int out_size, void* d_ws, size_t ws_size,
                              hipStream_t stream){
    (void)in_sizes; (void)n_in; (void)out_size; (void)ws_size;
    const float* src_feat = (const float*)d_in[0];
    const float* dst_feat = (const float*)d_in[1];
    const int*   edge_src = (const int*)d_in[2];
    const int*   edge_dst = (const int*)d_in[3];
    const int*   edge_typ = (const int*)d_in[4];
    const float* w1_w  = (const float*)d_in[5];
    const float* w1_b  = (const float*)d_in[6];
    const float* w2s_w = (const float*)d_in[7];
    const float* w2s_b = (const float*)d_in[8];
    const float* w2d_w = (const float*)d_in[9];
    const float* w2d_b = (const float*)d_in[10];
    const float* w4_w  = (const float*)d_in[11];
    const float* w4_b  = (const float*)d_in[12];
    const float* attn  = (const float*)d_in[13];
    const float* rel_emb = (const float*)d_in[14];
    float* out = (float*)d_out;

    // Workspace carve-up (all chunks 256B-aligned by construction).
    char* w = (char*)d_ws;
    _Float16* fs = (_Float16*)w; w += (size_t)N_SRC * 512 * sizeof(_Float16);   // 102.4 MB
    _Float16* fd = (_Float16*)w; w += (size_t)N_DST * 512 * sizeof(_Float16);   //  51.2 MB
    float*    exlog = (float*)w; w += (size_t)N_EDGES * NH * sizeof(float);     //  51.2 MB
    unsigned* segmax = (unsigned*)w; w += (size_t)N_DST * NH * sizeof(unsigned);
    float*    segsum = (float*)w;    w += (size_t)N_DST * NH * sizeof(float);
    float*    g = (float*)w;                                                    //  25.6 MB

    dim3 blk(256);
    gat_init   <<<(N_DST * 128 + 255) / 256, blk, 0, stream>>>(g, segsum, segmax);
    gat_gemm512<<<(N_SRC / 16) * 32 / 8, blk, 0, stream>>>(src_feat, w2s_w, w2s_b, fs, N_SRC);
    gat_gemm512<<<(N_DST / 16) * 32 / 8, blk, 0, stream>>>(dst_feat, w2d_w, w2d_b, fd, N_DST);
    gat_logits <<<N_EDGES / 8, blk, 0, stream>>>(fs, fd, edge_src, edge_dst, attn, exlog, segmax);
    gat_expsum <<<N_EDGES * NH / 256, blk, 0, stream>>>(edge_dst, segmax, exlog, segsum);
    gat_message<<<N_EDGES / 8, blk, 0, stream>>>(src_feat, rel_emb, edge_src, edge_dst, edge_typ,
                                                 exlog, segsum, g);
    gat_final  <<<(N_DST / 16) * 8 / 8, blk, 0, stream>>>(dst_feat, g, w1_w, w1_b, w4_w, w4_b, out);
}
